// M3GAT_14078902796502
// MI455X (gfx1250) — compile-verified, matrix operand order checked by reading, not run on hardware
//
#include <hip/hip_runtime.h>
#include <hip/hip_bf16.h>

// ---------------------------------------------------------------------------
// GAT (dense adjacency) forward for MI455X / gfx1250, wave32, WMMA bf16,
// CDNA5 async global->LDS staging with LDS double-buffering:
//   per K-step: WMMA from buffer[cur] while GLOBAL_LOAD_ASYNC_TO_LDS_B128
//   fills buffer[nxt]; one S_WAIT_ASYNCCNT + one barrier per step.
//   * adjacency -> packed bitmask (2MB, L2-resident)
//   * projections (X@W) via bf16 WMMA tiled GEMM (f32 accumulate)
//   * masked softmax: streaming online max/sum per row (no NxN matrix in HBM)
//   * aggregation: P tile (exp(e-m)/s) built in LDS, WMMA P @ Wh,
//     fused ELU / residual epilogue.
// ---------------------------------------------------------------------------

constexpr int NN      = 4096;
constexpr int NFEAT   = 1024;
constexpr int NHID    = 256;
constexpr int NHEADS  = 4;
constexpr int NCLASS  = 1024;
constexpr int NLAYERS = 2;
constexpr int MWORDS  = NN / 32;   // mask words per row

typedef __attribute__((ext_vector_type(16))) __bf16 v16bf;
typedef __attribute__((ext_vector_type(8)))  float  v8f;

union AFrag { v16bf v; unsigned u[8]; };

// float -> bf16 round-to-nearest-even via bit math (no native cast dependence)
__device__ inline __bf16 f2bf(float f) {
    union { unsigned u; float f; } c; c.f = f;
    unsigned r = ((c.u >> 16) & 1u) + 0x7fffu;
    unsigned short h = (unsigned short)((c.u + r) >> 16);
    union { unsigned short s; __bf16 b; } o; o.s = h;
    return o.b;
}

// K-pair base for 16-bit A fragment (ISA 7.12.2, 16-bit A 16x32):
// lanes 0-15: V0..V3 hold K=0..7, V4..V7 hold K=16..23
// lanes16-31: V0..V3 hold K=8..15, V4..V7 hold K=24..31
__device__ inline int kpb(int v, int laneHi) {
    return ((v & 4) << 2) + (laneHi << 3) + ((v & 3) << 1);
}

// LDS byte offset of a generic pointer to a __shared__ object.
// (ISA 10.2: shared-aperture flat address low 32 bits == LDS address)
__device__ inline unsigned lds_off(const void* p) {
    return (unsigned)(size_t)p;
}

// CDNA5 async copy: 16 bytes global -> LDS per lane (tracked by ASYNCcnt)
__device__ inline void async_copy_b128(unsigned lds_byte_off, const void* gptr) {
    asm volatile("global_load_async_to_lds_b128 %0, %1, off"
                 :: "v"(lds_byte_off), "v"((unsigned long long)(size_t)gptr)
                 : "memory");
}
__device__ inline void wait_asynccnt0() {
    asm volatile("s_wait_asynccnt 0" ::: "memory");
}

// ---------------------------------------------------------------------------
// Kernel 0: copy x -> x_cur (f32) and x_bf (bf16)
__global__ __launch_bounds__(256)
void init_x_kernel(const float* __restrict__ x, float* __restrict__ xc,
                   __bf16* __restrict__ xb, int n) {
    int t = blockIdx.x * blockDim.x + threadIdx.x;
    if (t < n) { float v = x[t]; xc[t] = v; xb[t] = f2bf(v); }
}

// Kernel 1: pack adjacency (int32 0/1) into bitmask, 1 word per thread
__global__ __launch_bounds__(256)
void pack_mask_kernel(const int* __restrict__ adj, unsigned* __restrict__ mp, int nwords) {
    int t = blockIdx.x * blockDim.x + threadIdx.x;
    if (t >= nwords) return;
    long base = (long)t * 32;
    unsigned bits = 0;
#pragma unroll
    for (int i = 0; i < 32; ++i) bits |= (adj[base + i] > 0 ? 1u : 0u) << i;
    mp[t] = bits;
}

// ---------------------------------------------------------------------------
// Tiled bf16 WMMA GEMM: C[M,ldc] = A_bf16[M,lda] @ B_f32[Ktot,ldb]
// 64x64 tile per 128-thread (4-wave) workgroup, batched via blockIdx.z.
// A tile: async global->LDS b128, double-buffered. B tile: f32->bf16 cvt,
// stored transposed [col][k], double-buffered.
__global__ __launch_bounds__(128)
void gemm_bf16_kernel(const __bf16* __restrict__ A, int lda,
                      const float* __restrict__ B, int ldb, long Bstride,
                      float* __restrict__ Cf, __bf16* __restrict__ Cb,
                      int ldc, long Cstride, int Ktot) {
    __shared__ __bf16 As[2][64 * 32];   // [buf][row][k]
    __shared__ __bf16 Bs[2][64 * 32];   // [buf][col][k]
    const int t = threadIdx.x;
    const int lane = t & 31, w = t >> 5;
    const int laneLo = lane & 15, laneHi = lane >> 4;
    const int n0 = blockIdx.x * 64, m0 = blockIdx.y * 64;
    const float* Bp = B + (long)blockIdx.z * Bstride;
    float*  Cfp = Cf + (long)blockIdx.z * Cstride;
    __bf16* Cbp = Cb + (long)blockIdx.z * Cstride;
    const unsigned abase = lds_off(&As[0][0]);

    // fixed-trip staging (t in [0,128): no thread-dependent bounds -> no exec masks)
    auto stageA = [&](int step, int buf) {
#pragma unroll
        for (int it = 0; it < 2; ++it) {
            int idx = t + it * 128;            // 0..255
            int r = idx >> 2, ci = idx & 3;
            async_copy_b128(abase + (unsigned)(buf * 4096 + (r * 32 + ci * 8) * 2),
                            A + (long)(m0 + r) * lda + step * 32 + ci * 8);
        }
    };
    auto stageB = [&](int step, int buf) {
#pragma unroll
        for (int it = 0; it < 16; ++it) {
            int idx = t + it * 128;            // 0..2047
            int k = idx >> 6, cc = idx & 63;
            Bs[buf][cc * 32 + k] = f2bf(Bp[(long)(step * 32 + k) * ldb + n0 + cc]);
        }
    };

    v8f acc[4];
#pragma unroll
    for (int i = 0; i < 4; ++i)
#pragma unroll
        for (int r = 0; r < 8; ++r) acc[i][r] = 0.f;

    const int NK = Ktot >> 5;
    stageA(0, 0);
    stageB(0, 0);
    wait_asynccnt0();
    __syncthreads();

    for (int s = 0; s < NK; ++s) {
        const int cur = s & 1, nxt = cur ^ 1;
        // fragment loads + WMMA from current buffer
        AFrag af;
#pragma unroll
        for (int v = 0; v < 8; ++v)
            af.u[v] = *reinterpret_cast<const unsigned*>(
                &As[cur][(w * 16 + laneLo) * 32 + kpb(v, laneHi)]);
#pragma unroll
        for (int nt = 0; nt < 4; ++nt) {
            AFrag bf;
#pragma unroll
            for (int v = 0; v < 8; ++v)
                bf.u[v] = *reinterpret_cast<const unsigned*>(
                    &Bs[cur][(nt * 16 + laneLo) * 32 + kpb(v, laneHi)]);
            acc[nt] = __builtin_amdgcn_wmma_f32_16x16x32_bf16(
                false, af.v, false, bf.v, (short)0, acc[nt], false, false);
        }
        // stage next tile into the other buffer (overlaps with WMMA)
        if (s + 1 < NK) {
            stageA(s + 1, nxt);
            stageB(s + 1, nxt);
        }
        wait_asynccnt0();
        __syncthreads();
    }
    // epilogue: C layout (VGPR r -> row r + 8*laneHi, col = laneLo)
#pragma unroll
    for (int nt = 0; nt < 4; ++nt) {
        int gcol = n0 + nt * 16 + laneLo;
#pragma unroll
        for (int r = 0; r < 8; ++r) {
            int grow = m0 + w * 16 + laneHi * 8 + r;
            float vv = acc[nt][r];
            Cfp[(long)grow * ldc + gcol] = vv;
            Cbp[(long)grow * ldc + gcol] = f2bf(vv);
        }
    }
}

// ---------------------------------------------------------------------------
// f_src / f_dst: one wave per row, dual dot product with shuffle reduce
__global__ __launch_bounds__(256)
void fsd_kernel(const float* __restrict__ Wh, const float* __restrict__ a,
                int aStride, int d, float* __restrict__ f_src,
                float* __restrict__ f_dst, int nRows) {
    int gw = (blockIdx.x * blockDim.x + threadIdx.x) >> 5;
    int lane = threadIdx.x & 31;
    if (gw >= nRows) return;
    int batch = gw >> 12;                       // NN == 4096
    const float* av  = a + (long)batch * aStride;
    const float* row = Wh + (long)gw * d;
    float fs = 0.f, fd = 0.f;
    for (int c = lane; c < d; c += 32) {
        float wv = row[c];
        fs += wv * av[c];
        fd += wv * av[d + c];
    }
#pragma unroll
    for (int off = 16; off > 0; off >>= 1) {
        fs += __shfl_down(fs, off, 32);
        fd += __shfl_down(fd, off, 32);
    }
    if (lane == 0) { f_src[gw] = fs; f_dst[gw] = fd; }
}

// ---------------------------------------------------------------------------
// Online masked-softmax stats: one wave per row -> rowmax, 1/rowsum
__global__ __launch_bounds__(256)
void stats_kernel(const unsigned* __restrict__ mp,
                  const float* __restrict__ f_src, const float* __restrict__ f_dst,
                  float* __restrict__ rowmax, float* __restrict__ rowrcp, int nRows) {
    int gw = (blockIdx.x * blockDim.x + threadIdx.x) >> 5;
    int lane = threadIdx.x & 31;
    if (gw >= nRows) return;
    int batch = gw >> 12;
    int r = gw & (NN - 1);
    float fs = f_src[gw];
    const float* fd = f_dst + (long)batch * NN;
    float m = -3.0e38f, s = 0.f;
    for (int base = 0; base < NN; base += 32) {
        unsigned word = mp[(long)r * MWORDS + (base >> 5)];   // uniform per wave
        if ((word >> lane) & 1u) {
            float tv = fs + fd[base + lane];
            float e = fmaxf(tv, 0.2f * tv);                   // leaky relu
            if (e > m) { s = s * __expf(m - e) + 1.0f; m = e; }
            else        s += __expf(e - m);
        }
    }
#pragma unroll
    for (int off = 16; off > 0; off >>= 1) {
        float mo = __shfl_xor(m, off, 32);
        float so = __shfl_xor(s, off, 32);
        float M  = fmaxf(m, mo);
        s = s * __expf(m - M) + so * __expf(mo - M);
        m = M;
    }
    if (lane == 0) { rowmax[gw] = m; rowrcp[gw] = 1.0f / s; }
}

// ---------------------------------------------------------------------------
// Fused aggregation: out = ELU( softmax_masked(e) @ V ) [+ residual]
// blockIdx.x = head (heads mode) or 256-col chunk (out mode); blockIdx.y = row block
// V tile async-staged row-major [k][col], double-buffered; P tile in LDS [row][k].
__global__ __launch_bounds__(128)
void att_aggr_kernel(const unsigned* __restrict__ mp,
                     const float* __restrict__ f_src, const float* __restrict__ f_dst,
                     const float* __restrict__ rowmax, const float* __restrict__ rowrcp,
                     const __bf16* __restrict__ V, int ldv,
                     long statStride, long VStride, int colStride, int outColStride,
                     __bf16* __restrict__ outB, int ldo,
                     const float* __restrict__ resid, float* __restrict__ outF) {
    __shared__ __bf16 Ps[2][16 * 32];    // [buf][row][k]
    __shared__ __bf16 Vs[2][32 * 256];   // [buf][k][col] (async target)
    const int t = threadIdx.x, lane = t & 31, w = t >> 5;
    const int laneLo = lane & 15, laneHi = lane >> 4;
    const int c = blockIdx.x;
    const int row0 = blockIdx.y * 16;
    const long sOff = (long)c * statStride;
    const __bf16* Vp = V + (long)c * VStride;
    const int colbase = c * colStride;
    const int outcol0 = c * outColStride;
    const unsigned vbase = lds_off(&Vs[0][0]);

    const int prow = t >> 3;          // 0..15 : P row this thread fills
    const int pk   = (t & 7) * 4;     // 4 consecutive k values
    const float fs = f_src[sOff + row0 + prow];
    const float m  = rowmax[sOff + row0 + prow];
    const float rs = rowrcp[sOff + row0 + prow];

    auto stageV = [&](int step, int buf) {
#pragma unroll
        for (int it = 0; it < 8; ++it) {
            int idx = t + it * 128;           // 0..1023
            int k = idx >> 5, ci = idx & 31;
            async_copy_b128(vbase + (unsigned)(buf * 16384 + (k * 256 + ci * 8) * 2),
                            Vp + (long)(step * 32 + k) * ldv + colbase + ci * 8);
        }
    };
    auto stageP = [&](int step, int buf) {
        unsigned mw = mp[(long)(row0 + prow) * MWORDS + step];
#pragma unroll
        for (int i = 0; i < 4; ++i) {
            float p = 0.f;
            if ((mw >> (pk + i)) & 1u) {
                float tv = fs + f_dst[sOff + step * 32 + pk + i];
                float e = fmaxf(tv, 0.2f * tv);
                p = __expf(e - m) * rs;
            }
            Ps[buf][prow * 32 + pk + i] = f2bf(p);
        }
    };

    v8f acc[4];
#pragma unroll
    for (int i = 0; i < 4; ++i)
#pragma unroll
        for (int r = 0; r < 8; ++r) acc[i][r] = 0.f;

    const int NK = NN / 32;   // 128 K-steps
    stageV(0, 0);
    stageP(0, 0);
    wait_asynccnt0();
    __syncthreads();

    for (int s = 0; s < NK; ++s) {
        const int cur = s & 1, nxt = cur ^ 1;
        // fragment loads + WMMA from current buffers
        AFrag af;
#pragma unroll
        for (int v = 0; v < 8; ++v)
            af.u[v] = *reinterpret_cast<const unsigned*>(
                &Ps[cur][laneLo * 32 + kpb(v, laneHi)]);
#pragma unroll
        for (int nt = 0; nt < 4; ++nt) {
            int col = w * 64 + nt * 16 + laneLo;
            AFrag bf;
#pragma unroll
            for (int v = 0; v < 8; ++v) {
                int kb = kpb(v, laneHi);
                unsigned lo = *reinterpret_cast<const unsigned short*>(&Vs[cur][kb * 256 + col]);
                unsigned hi = *reinterpret_cast<const unsigned short*>(&Vs[cur][(kb + 1) * 256 + col]);
                bf.u[v] = lo | (hi << 16);
            }
            acc[nt] = __builtin_amdgcn_wmma_f32_16x16x32_bf16(
                false, af.v, false, bf.v, (short)0, acc[nt], false, false);
        }
        // stage next tile into the other buffer (overlaps with WMMA)
        if (s + 1 < NK) {
            stageV(s + 1, nxt);
            stageP(s + 1, nxt);
        }
        wait_asynccnt0();
        __syncthreads();
    }
    // epilogue: ELU, optional residual + f32 out
#pragma unroll
    for (int nt = 0; nt < 4; ++nt) {
        int gcol = outcol0 + w * 64 + nt * 16 + laneLo;
#pragma unroll
        for (int r = 0; r < 8; ++r) {
            int grow = row0 + laneHi * 8 + r;
            float vv = acc[nt][r];
            float e = (vv > 0.f) ? vv : (__expf(vv) - 1.f);
            if (outF) {
                float xo = resid[(long)grow * ldo + gcol] + e;
                outF[(long)grow * ldo + gcol] = xo;
                outB[(long)grow * ldo + gcol] = f2bf(xo);
            } else {
                outB[(long)grow * ldo + gcol] = f2bf(e);
            }
        }
    }
}

// ---------------------------------------------------------------------------
extern "C" void kernel_launch(void* const* d_in, const int* in_sizes, int n_in,
                              void* d_out, int out_size, void* d_ws, size_t ws_size,
                              hipStream_t stream) {
    const float* x       = (const float*)d_in[0];
    const int*   adj     = (const int*)d_in[1];
    const float* W_heads = (const float*)d_in[2];
    const float* a_heads = (const float*)d_in[3];
    const float* W_out   = (const float*)d_in[4];
    const float* a_out   = (const float*)d_in[5];
    float* out = (float*)d_out;

    // workspace layout (~58.3 MB total)
    char* ws = (char*)d_ws;
    size_t o = 0;
    float*    X_CUR  = (float*)(ws + o);    o += (size_t)NN * NFEAT * 4;          // 16MB
    __bf16*   X_BF   = (__bf16*)(ws + o);   o += (size_t)NN * NFEAT * 2;          // 8MB
    float*    WH_F   = (float*)(ws + o);    o += (size_t)NHEADS * NN * NHID * 4;  // 16MB (reused for out: NN*1024)
    __bf16*   WH_B   = (__bf16*)(ws + o);   o += (size_t)NHEADS * NN * NHID * 2;  // 8MB
    __bf16*   HCAT_B = (__bf16*)(ws + o);   o += (size_t)NN * NFEAT * 2;          // 8MB
    unsigned* MASKP  = (unsigned*)(ws + o); o += (size_t)NN * MWORDS * 4;         // 2MB
    float*    F_SRC  = (float*)(ws + o);    o += (size_t)NHEADS * NN * 4;
    float*    F_DST  = (float*)(ws + o);    o += (size_t)NHEADS * NN * 4;
    float*    R_MAX  = (float*)(ws + o);    o += (size_t)NHEADS * NN * 4;
    float*    R_RCP  = (float*)(ws + o);    o += (size_t)NHEADS * NN * 4;

    init_x_kernel<<<(NN * NFEAT + 255) / 256, 256, 0, stream>>>(x, X_CUR, X_BF, NN * NFEAT);
    pack_mask_kernel<<<(NN * MWORDS + 255) / 256, 256, 0, stream>>>(adj, MASKP, NN * MWORDS);

    for (int l = 0; l < NLAYERS; ++l) {
        const float* Wh_l = W_heads + (size_t)l * NHEADS * NFEAT * NHID;
        const float* ah_l = a_heads + (size_t)l * NHEADS * 2 * NHID;
        const float* Wo_l = W_out   + (size_t)l * (NHID * NHEADS) * NCLASS;
        const float* ao_l = a_out   + (size_t)l * 2 * NCLASS;

        // ---- heads: Wh = X @ W  (batched over 4 heads) ----
        gemm_bf16_kernel<<<dim3(NHID / 64, NN / 64, NHEADS), 128, 0, stream>>>(
            X_BF, NFEAT, Wh_l, NHID, (long)NFEAT * NHID,
            WH_F, WH_B, NHID, (long)NN * NHID, NFEAT);
        fsd_kernel<<<(NHEADS * NN) / 8, 256, 0, stream>>>(
            WH_F, ah_l, 2 * NHID, NHID, F_SRC, F_DST, NHEADS * NN);
        stats_kernel<<<(NHEADS * NN) / 8, 256, 0, stream>>>(
            MASKP, F_SRC, F_DST, R_MAX, R_RCP, NHEADS * NN);
        // heads aggregation -> HCAT (bf16, concat across heads), fused ELU
        att_aggr_kernel<<<dim3(NHEADS, NN / 16), 128, 0, stream>>>(
            MASKP, F_SRC, F_DST, R_MAX, R_RCP,
            WH_B, NHID, (long)NN, (long)NN * NHID, /*colStride=*/0, /*outColStride=*/NHID,
            HCAT_B, NFEAT, nullptr, nullptr);

        // ---- out layer: Wh_out = HCAT @ W_out ----
        gemm_bf16_kernel<<<dim3(NCLASS / 64, NN / 64, 1), 128, 0, stream>>>(
            HCAT_B, NFEAT, Wo_l, NCLASS, 0,
            WH_F, WH_B, NCLASS, 0, NFEAT);
        fsd_kernel<<<NN / 8, 256, 0, stream>>>(
            WH_F, ao_l, 0, NCLASS, F_SRC, F_DST, NN);
        stats_kernel<<<NN / 8, 256, 0, stream>>>(
            MASKP, F_SRC, F_DST, R_MAX, R_RCP, NN);
        // out aggregation with residual: x_new = x_in + elu(att @ Wh_out)
        float* xdstF = (l == NLAYERS - 1) ? out : X_CUR;   // in-place safe (1:1 rd/wr per thread)
        att_aggr_kernel<<<dim3(NCLASS / 256, NN / 16), 128, 0, stream>>>(
            MASKP, F_SRC, F_DST, R_MAX, R_RCP,
            WH_B, NCLASS, 0, 0, /*colStride=*/256, /*outColStride=*/256,
            X_BF, NCLASS, X_CUR, xdstF);
    }
}
